// SnailFewShot_86517821214906
// MI455X (gfx1250) — matrix operand order repaired and checked by reading, not compiled, not run on hardware
//
#include <hip/hip_runtime.h>
#include <hip/hip_bf16.h>
#include <math.h>

typedef __attribute__((ext_vector_type(16))) _Float16 v16h;
typedef __attribute__((ext_vector_type(8)))  _Float16 v8h;
typedef __attribute__((ext_vector_type(8)))  float    v8f;

static constexpr int TOTAL = 1664;   // 64 episodes * 26 steps
static constexpr int SEQL  = 26;
static constexpr int NEP   = 64;
static constexpr int PARTS = 256;    // partial-stat slots for BN reduction

__host__ __device__ constexpr int al32(int k) { return (k + 31) & ~31; }

#define SHUF16(a, b) __builtin_shufflevector(a, b, 0,1,2,3,4,5,6,7,8,9,10,11,12,13,14,15)

// ---------------------------------------------------------------------------
// utility
// ---------------------------------------------------------------------------
__global__ void k_zero(float* __restrict__ p, long long n) {
  long long i = (long long)blockIdx.x * blockDim.x + threadIdx.x;
  if (i < n) p[i] = 0.f;
}

// f32 [rows][C] -> f16 [rows][Cp] (zero-padded); also used for weight matrices
__global__ void cvt_pad_f16(const float* __restrict__ src, _Float16* __restrict__ dst,
                            long long rows, int C, int Cp) {
  const long long total = rows * Cp;
  const long long i = (long long)blockIdx.x * blockDim.x + threadIdx.x;
  if (i >= total) return;
  const long long r = i / Cp;
  const int c = (int)(i - r * Cp);
  dst[i] = (c < C) ? (_Float16)src[r * C + c] : (_Float16)0.f;
}

// ---------------------------------------------------------------------------
// Encoder convs: implicit GEMM, tap-major K ordering (k = tap*CIN + cin).
// Weights repacked once: wrep[n][tap*CIN+cin] f16, rows padded to Kp.
// ---------------------------------------------------------------------------
template <int CIN>
__global__ void conv_wrepack(const float* __restrict__ w, _Float16* __restrict__ out) {
  constexpr int K = CIN * 9, Kp = al32(K);
  const int i = blockIdx.x * blockDim.x + threadIdx.x;
  if (i >= 64 * Kp) return;
  const int nch = i / Kp, k = i - nch * Kp;
  _Float16 v = (_Float16)0.f;
  if (k < K) {
    const int tap = k / CIN, cin = k - tap * CIN;
    v = (_Float16)w[((long long)nch * CIN + cin) * 9 + tap];
  }
  out[i] = v;
}

// One wave -> 16-position (4x4 spatial) x 64-channel conv3x3(pad=1) tile.
template <int CIN, int H, int W>
__device__ __forceinline__ void conv_tile_acc_t(
    const float* __restrict__ img, const _Float16* __restrict__ wrep,
    int n, int by, int bx, int lane, v8f* acc)
{
  constexpr int K = CIN * 9, Kp = al32(K);
  constexpr bool NEEDP = (H % 4 != 0) || (W % 4 != 0);
  const int hf = lane >> 4, l15 = lane & 15;
  const int py = l15 >> 2, px = l15 & 3;
  const int h = by * 4 + py, w = bx * 4 + px;
  const bool posV = NEEDP ? ((h < H) && (w < W)) : true;
  const float* ibase = img + (long long)n * CIN * H * W;

  for (int kb = 0; kb < Kp; kb += 32) {
    v16h af;
    if constexpr (CIN == 1) {               // K=9: single padded step
#pragma unroll
      for (int j = 0; j < 16; ++j) {
        const int k = kb + ((j < 8) ? (j + hf * 8) : (j + 8 + hf * 8));
        float v = 0.f;
        if (posV && k < K) {
          const int ky = k / 3, kx = k - ky * 3;
          const int ih = h + ky - 1, iw = w + kx - 1;
          if (ih >= 0 && ih < H && iw >= 0 && iw < W) v = ibase[ih * W + iw];
        }
        af[j] = (_Float16)v;
      }
    } else {                                 // CIN mult of 8: tap const per 8-group
#pragma unroll
      for (int g = 0; g < 2; ++g) {
        const int k0 = kb + (g ? 16 : 0) + hf * 8;  // multiple of 8
        const int tap = k0 / CIN, cin0 = k0 - tap * CIN;
        const int ky = tap / 3, kx = tap - ky * 3;
        const int ih = h + ky - 1, iw = w + kx - 1;
        const bool v2 = posV && ih >= 0 && ih < H && iw >= 0 && iw < W;
        const float* p = ibase + (long long)cin0 * (H * W) + ih * W + iw;
#pragma unroll
        for (int e = 0; e < 8; ++e)
          af[g * 8 + e] = (_Float16)(v2 ? p[e * (H * W)] : 0.f);
      }
    }
#pragma unroll
    for (int t = 0; t < 4; ++t) {
      const _Float16* bp = wrep + (long long)(t * 16 + l15) * Kp + kb + hf * 8;
      v8h b0 = *(const v8h*)bp;
      v8h b1 = *(const v8h*)(bp + 16);
      v16h bf = SHUF16(b0, b1);
      acc[t] = __builtin_amdgcn_wmma_f32_16x16x32_f16(
          false, af, false, bf, (short)0, acc[t], false, false);
    }
  }
}

// Pass 1: conv + per-channel sum / sumsq partials (slotted atomics).
template <int CIN, int H, int W>
__global__ void conv_stats_t(const float* __restrict__ img,
                             const _Float16* __restrict__ wrep,
                             float* __restrict__ partial)
{
  constexpr int tilesX = (W + 3) / 4;
  const int n = blockIdx.y;
  const int by = blockIdx.x / tilesX, bx = blockIdx.x - by * tilesX;
  const int lane = threadIdx.x;
  v8f acc[4] = {};
  conv_tile_acc_t<CIN, H, W>(img, wrep, n, by, bx, lane, acc);

  const int hf = lane >> 4, l15 = lane & 15;
  const int slot = (blockIdx.y * gridDim.x + blockIdx.x) & (PARTS - 1);
#pragma unroll
  for (int t = 0; t < 4; ++t) {
    float s = 0.f, q = 0.f;
#pragma unroll
    for (int r = 0; r < 8; ++r) { const float v = acc[t][r]; s += v; q += v * v; }
    s += __shfl_down(s, 16);
    q += __shfl_down(q, 16);
    if (hf == 0) {
      const int c = t * 16 + l15;
      atomicAdd(&partial[((long long)slot * 64 + c) * 2 + 0], s);
      atomicAdd(&partial[((long long)slot * 64 + c) * 2 + 1], q);
    }
  }
}

// Fold BN (batch stats, biased var, eps 1e-5) + gamma/beta into scale/shift.
// (conv bias provably cancels inside training-mode BN, so it is skipped.)
__global__ void bn_finalize_kernel(const float* __restrict__ partial,
                                   const float* __restrict__ g,
                                   const float* __restrict__ beta,
                                   float* __restrict__ ss, float cntInv)
{
  const int c = threadIdx.x;  // 64 channels
  float s = 0.f, q = 0.f;
  for (int p = 0; p < PARTS; ++p) {
    s += partial[((long long)p * 64 + c) * 2 + 0];
    q += partial[((long long)p * 64 + c) * 2 + 1];
  }
  const float mean = s * cntInv;
  const float var  = q * cntInv - mean * mean;
  const float inv  = rsqrtf(var + 1e-5f);
  const float sc   = g[c] * inv;
  ss[c]      = sc;
  ss[64 + c] = beta[c] - mean * sc;
}

// Pass 2: recompute conv, scale/shift + relu into LDS, maxpool2 from LDS.
template <int CIN, int H, int W>
__global__ void conv_apply_t(const float* __restrict__ img,
                             const _Float16* __restrict__ wrep,
                             const float* __restrict__ ss,
                             float* __restrict__ out)
{
  __shared__ float tile[16 * 64];
  constexpr int tilesX = (W + 3) / 4;
  const int n = blockIdx.y;
  const int by = blockIdx.x / tilesX, bx = blockIdx.x - by * tilesX;
  const int lane = threadIdx.x;
  v8f acc[4] = {};
  conv_tile_acc_t<CIN, H, W>(img, wrep, n, by, bx, lane, acc);

  const int hf = lane >> 4, l15 = lane & 15;
#pragma unroll
  for (int t = 0; t < 4; ++t) {
    const int c = t * 16 + l15;
    const float scl = ss[c], sh = ss[64 + c];
#pragma unroll
    for (int r = 0; r < 8; ++r)
      tile[(r + 8 * hf) * 64 + c] = fmaxf(acc[t][r] * scl + sh, 0.f);
  }
  __syncthreads();

  constexpr int PH = H / 2, PW = W / 2;
#pragma unroll
  for (int it = 0; it < 8; ++it) {
    const int o = lane + 32 * it;           // 4 pool cells x 64 ch
    const int c = o & 63;
    const int cell = o >> 6;
    const int i = cell >> 1, j = cell & 1;
    const int gy = by * 2 + i, gx = bx * 2 + j;
    if (gy < PH && gx < PW) {
      const int m0 = (2 * i) * 4 + 2 * j;
      const int m2 = (2 * i + 1) * 4 + 2 * j;
      const float mv = fmaxf(fmaxf(tile[m0 * 64 + c], tile[(m0 + 1) * 64 + c]),
                             fmaxf(tile[m2 * 64 + c], tile[(m2 + 1) * 64 + c]));
      out[(((long long)n * 64 + c) * PH + gy) * PW + gx] = mv;
    }
  }
}

// ---------------------------------------------------------------------------
// Generic WMMA GEMM: C[M,N] = A[M,Kp]_f16 * B[N,Kp]_f16^T + bias.
// Kp multiple of 32, M multiple of 16 (always TOTAL). One wave -> 16x64 tile.
// Each fragment = two b128 loads; no per-element predication.
// ---------------------------------------------------------------------------
__global__ void gemm_wmma_f16(const _Float16* __restrict__ A, const _Float16* __restrict__ B,
                              const float* __restrict__ bias, float* __restrict__ C,
                              int N, int Kp)
{
  const int mtile = blockIdx.y * 16;
  const int nbase = blockIdx.x * 64;
  const int lane = threadIdx.x;
  const int hf = lane >> 4, l15 = lane & 15;
  const _Float16* aPtr = A + (long long)(mtile + l15) * Kp;
  const _Float16* bP0 = B + (long long)((nbase +      l15) < N ? (nbase +      l15) : 0) * Kp;
  const _Float16* bP1 = B + (long long)((nbase + 16 + l15) < N ? (nbase + 16 + l15) : 0) * Kp;
  const _Float16* bP2 = B + (long long)((nbase + 32 + l15) < N ? (nbase + 32 + l15) : 0) * Kp;
  const _Float16* bP3 = B + (long long)((nbase + 48 + l15) < N ? (nbase + 48 + l15) : 0) * Kp;

  v8f acc0 = {}, acc1 = {}, acc2 = {}, acc3 = {};
  for (int kb = 0; kb < Kp; kb += 32) {
    v8h a0 = *(const v8h*)(aPtr + kb + hf * 16);
    v8h a1 = *(const v8h*)(aPtr + kb + hf * 16 + 8);
    v16h af = SHUF16(a0, a1);
    const int kbb = kb + hf * 16;
    {
      v16h bf = SHUF16(*(const v8h*)(bP0 + kbb), *(const v8h*)(bP0 + kbb + 8));
      acc0 = __builtin_amdgcn_wmma_f32_16x16x32_f16(false, af, false, bf, (short)0, acc0, false, false);
    }
    {
      v16h bf = SHUF16(*(const v8h*)(bP1 + kbb), *(const v8h*)(bP1 + kbb + 8));
      acc1 = __builtin_amdgcn_wmma_f32_16x16x32_f16(false, af, false, bf, (short)0, acc1, false, false);
    }
    {
      v16h bf = SHUF16(*(const v8h*)(bP2 + kbb), *(const v8h*)(bP2 + kbb + 8));
      acc2 = __builtin_amdgcn_wmma_f32_16x16x32_f16(false, af, false, bf, (short)0, acc2, false, false);
    }
    {
      v16h bf = SHUF16(*(const v8h*)(bP3 + kbb), *(const v8h*)(bP3 + kbb + 8));
      acc3 = __builtin_amdgcn_wmma_f32_16x16x32_f16(false, af, false, bf, (short)0, acc3, false, false);
    }
  }
  v8f accs[4] = {acc0, acc1, acc2, acc3};
#pragma unroll
  for (int t = 0; t < 4; ++t) {
    const int col = nbase + t * 16 + l15;
    if (col < N) {
      const float bv = bias ? bias[col] : 0.f;
#pragma unroll
      for (int r = 0; r < 8; ++r)
        C[(long long)(mtile + r + 8 * hf) * N + col] = accs[t][r] + bv;
    }
  }
}

// ---------------------------------------------------------------------------
// SNAIL glue kernels
// ---------------------------------------------------------------------------
__global__ void build_x0(const float* __restrict__ feat, const float* __restrict__ labels,
                         float* __restrict__ x0)
{
  const long long i = (long long)blockIdx.x * blockDim.x + threadIdx.x;
  if (i >= (long long)TOTAL * 69) return;
  const int row = (int)(i / 69), c = (int)(i % 69);
  if (c < 64) x0[i] = feat[(long long)row * 64 + c];
  else {
    const bool last = (row % SEQL) == (SEQL - 1);
    x0[i] = last ? 0.f : labels[(long long)row * 5 + (c - 64)];
  }
}

// scores[b][q][k] = <q_row,k_row>*invs, masked (k>q) with -1e30
__global__ void attn_scores(const float* __restrict__ q, const float* __restrict__ k,
                            float* __restrict__ scores, int ksz, float invs)
{
  const int b = blockIdx.x;
  for (int p = threadIdx.x; p < SEQL * SEQL; p += blockDim.x) {
    const int qi = p / SEQL, ki = p - qi * SEQL;
    float s;
    if (ki > qi) s = -1e30f;
    else {
      s = 0.f;
      const float* qp = q + ((long long)b * SEQL + qi) * ksz;
      const float* kp = k + ((long long)b * SEQL + ki) * ksz;
      for (int c = 0; c < ksz; ++c) s += qp[c] * kp[c];
      s *= invs;
    }
    scores[((long long)b * SEQL + qi) * SEQL + ki] = s;
  }
}

// softmax over the QUERY dim (axis=1), faithful to the reference quirk
__global__ void attn_softmax_q(const float* __restrict__ scores, float* __restrict__ attn)
{
  const int idx = blockIdx.x * blockDim.x + threadIdx.x;  // b*SEQL + ki
  if (idx >= NEP * SEQL) return;
  const int b = idx / SEQL, ki = idx - b * SEQL;
  float mx = -1e30f;
  for (int qi = 0; qi < SEQL; ++qi)
    mx = fmaxf(mx, scores[((long long)b * SEQL + qi) * SEQL + ki]);
  float sum = 0.f;
  for (int qi = 0; qi < SEQL; ++qi)
    sum += __expf(scores[((long long)b * SEQL + qi) * SEQL + ki] - mx);
  const float inv = 1.f / sum;
  for (int qi = 0; qi < SEQL; ++qi) {
    const long long o = ((long long)b * SEQL + qi) * SEQL + ki;
    attn[o] = __expf(scores[o] - mx) * inv;
  }
}

// xout[:, :C] = x ; xout[:, C:] = attn @ v
__global__ void attn_out_concat(const float* __restrict__ x, const float* __restrict__ attn,
                                const float* __restrict__ v, float* __restrict__ xout,
                                int C, int vsz)
{
  const int CT = C + vsz;
  const long long i = (long long)blockIdx.x * blockDim.x + threadIdx.x;
  if (i >= (long long)TOTAL * CT) return;
  const int row = (int)(i / CT), c = (int)(i % CT);
  if (c < C) { xout[i] = x[(long long)row * C + c]; return; }
  const int d = c - C;
  const int b = row / SEQL, qi = row - b * SEQL;
  float s = 0.f;
  for (int ki = 0; ki < SEQL; ++ki)
    s += attn[((long long)b * SEQL + qi) * SEQL + ki] *
         v[((long long)b * SEQL + ki) * vsz + d];
  xout[i] = s;
}

// causal conv (ksz=2, dil d) rows: Ag[b,t] = [x(t-d), x(t)] as f16, K padded
__global__ void tc_gather_f16(const float* __restrict__ x, _Float16* __restrict__ Ag,
                              int C, int Kp2, int d)
{
  const long long i = (long long)blockIdx.x * blockDim.x + threadIdx.x;
  if (i >= (long long)TOTAL * Kp2) return;
  const int row = (int)(i / Kp2), k = (int)(i % Kp2);
  const int b = row / SEQL, t = row - b * SEQL;
  float v = 0.f;
  if (k < C)            v = (t >= d) ? x[((long long)b * SEQL + (t - d)) * C + k] : 0.f;
  else if (k < 2 * C)   v = x[(long long)row * C + (k - C)];
  Ag[i] = (_Float16)v;
}

// repack conv1d weight [128][C][2] -> Bt[128][Kp2] f16 with k = j*C + c
__global__ void tc_repack_f16(const float* __restrict__ w, _Float16* __restrict__ Bt,
                              int C, int Kp2)
{
  const long long i = (long long)blockIdx.x * blockDim.x + threadIdx.x;
  if (i >= (long long)128 * Kp2) return;
  const int f = (int)(i / Kp2), k = (int)(i % Kp2);
  float v = 0.f;
  if (k < 2 * C) {
    const int j = k / C, c = k - j * C;
    v = w[((long long)f * C + c) * 2 + j];
  }
  Bt[i] = (_Float16)v;
}

// xout[:, :C] = x ; xout[:, C:] = tanh(yf) * sigmoid(yg)
__global__ void tc_gate_concat(const float* __restrict__ x, const float* __restrict__ yf,
                               const float* __restrict__ yg, float* __restrict__ xout, int C)
{
  const int CT = C + 128;
  const long long i = (long long)blockIdx.x * blockDim.x + threadIdx.x;
  if (i >= (long long)TOTAL * CT) return;
  const int row = (int)(i / CT), c = (int)(i % CT);
  if (c < C) { xout[i] = x[(long long)row * C + c]; return; }
  const int f = c - C;
  const float a = tanhf(yf[(long long)row * 128 + f]);
  const float g = 1.f / (1.f + __expf(-yg[(long long)row * 128 + f]));
  xout[i] = a * g;
}

// ---------------------------------------------------------------------------
// host orchestration
// ---------------------------------------------------------------------------
template <int CIN, int H, int W>
static void run_conv_layer(const float* img, const float* w, const float* g,
                           const float* beta, float* out, float* partial, float* ssbuf,
                           _Float16* wrep, hipStream_t stream)
{
  constexpr int K = CIN * 9, Kp = al32(K);
  conv_wrepack<CIN><<<(64 * Kp + 255) / 256, 256, 0, stream>>>(w, wrep);
  {
    const long long n = (long long)PARTS * 64 * 2;
    k_zero<<<(int)((n + 255) / 256), 256, 0, stream>>>(partial, n);
  }
  constexpr int tilesX = (W + 3) / 4, tilesY = (H + 3) / 4;
  dim3 grid(tilesX * tilesY, TOTAL);
  conv_stats_t<CIN, H, W><<<grid, 32, 0, stream>>>(img, wrep, partial);
  bn_finalize_kernel<<<1, 64, 0, stream>>>(partial, g, beta, ssbuf,
                                           1.f / ((float)TOTAL * H * W));
  conv_apply_t<CIN, H, W><<<grid, 32, 0, stream>>>(img, wrep, ssbuf, out);
}

extern "C" void kernel_launch(void* const* d_in, const int* in_sizes, int n_in,
                              void* d_out, int out_size, void* d_ws, size_t ws_size,
                              hipStream_t stream)
{
  const float* P[80];
  for (int i = 0; i < n_in && i < 80; ++i) P[i] = (const float*)d_in[i];
  // layout: 0 input, 1 labels, 2..17 enc (w,b,g,beta)x4, 18..23 att1,
  //         24..43 tc1 (w1,b1,w2,b2)x5, 44..49 att2, 50..69 tc2,
  //         70..75 att3, 76 fc_w, 77 fc_b

  // ---- workspace bump allocator (bytes, 64B aligned) ----
  size_t off = 0;
  auto allocB = [&](size_t bytes) {
    char* p = (char*)d_ws + off;
    off += (bytes + 63) & ~(size_t)63;
    return (void*)p;
  };
  auto allocF = [&](long long nfl) { return (float*)allocB((size_t)nfl * 4); };
  auto allocH = [&](long long nh)  { return (_Float16*)allocB((size_t)nh * 2); };

  float* p1      = allocF((long long)TOTAL * 64 * 14 * 14);
  float* p2      = allocF((long long)TOTAL * 64 * 7 * 7);
  float* p3      = allocF((long long)TOTAL * 64 * 3 * 3);
  float* feat    = allocF((long long)TOTAL * 64);
  float* partial = allocF((long long)PARTS * 64 * 2);
  float* ssbuf   = allocF(128);
  float* x0      = allocF((long long)TOTAL * 69);
  float* qb      = allocF((long long)TOTAL * 512);
  float* kb      = allocF((long long)TOTAL * 512);
  float* vb      = allocF((long long)TOTAL * 256);
  float* scb     = allocF((long long)NEP * SEQL * SEQL);
  float* atb     = allocF((long long)NEP * SEQL * SEQL);
  float* xA      = allocF((long long)TOTAL * 1765);
  float* xB      = allocF((long long)TOTAL * 1765);
  float* yf      = allocF((long long)TOTAL * 128);
  float* yg      = allocF((long long)TOTAL * 128);
  _Float16* xh   = allocH((long long)TOTAL * 1792);   // f16 A for projections/fc
  _Float16* wbh  = allocH((long long)512 * 1792);     // f16 B (weights), reused
  _Float16* Agh  = allocH((long long)TOTAL * 2784);   // f16 A for tc gemms
  _Float16* Bt1h = allocH((long long)128 * 2784);
  _Float16* Bt2h = allocH((long long)128 * 2784);
  _Float16* wrep = allocH((long long)64 * 576);       // conv weights, reused

  auto gemm = [&](const _Float16* A, const _Float16* B, const float* bias, float* C,
                  int N, int Kp) {
    dim3 grid((N + 63) / 64, TOTAL / 16);
    gemm_wmma_f16<<<grid, 32, 0, stream>>>(A, B, bias, C, N, Kp);
  };
  auto cvtpad = [&](const float* src, _Float16* dst, long long rows, int C, int Cp) {
    const long long n = rows * Cp;
    cvt_pad_f16<<<(int)((n + 255) / 256), 256, 0, stream>>>(src, dst, rows, C, Cp);
  };

  // ---- encoder: 4x (conv3x3 -> BN(train) -> relu -> maxpool2) ----
  run_conv_layer<1, 28, 28>(P[0], P[2],  P[4],  P[5],  p1,   partial, ssbuf, wrep, stream);
  run_conv_layer<64, 14, 14>(p1,  P[6],  P[8],  P[9],  p2,   partial, ssbuf, wrep, stream);
  run_conv_layer<64, 7, 7>(p2,   P[10], P[12], P[13], p3,   partial, ssbuf, wrep, stream);
  run_conv_layer<64, 3, 3>(p3,   P[14], P[16], P[17], feat, partial, ssbuf, wrep, stream);

  // ---- x0 = concat(features, labels-with-query-zeroed) ----
  {
    const long long n = (long long)TOTAL * 69;
    build_x0<<<(int)((n + 255) / 256), 256, 0, stream>>>(feat, P[1], x0);
  }

  // ---- attention block ----
  auto attention = [&](const float* xin, int C, int ksz, int vsz, int pidx, float* xout) {
    const int Cp = al32(C);
    cvtpad(xin, xh, TOTAL, C, Cp);
    cvtpad(P[pidx + 0], wbh, ksz, C, Cp);
    gemm(xh, wbh, P[pidx + 1], qb, ksz, Cp);
    cvtpad(P[pidx + 2], wbh, ksz, C, Cp);
    gemm(xh, wbh, P[pidx + 3], kb, ksz, Cp);
    cvtpad(P[pidx + 4], wbh, vsz, C, Cp);
    gemm(xh, wbh, P[pidx + 5], vb, vsz, Cp);
    attn_scores<<<NEP, 256, 0, stream>>>(qb, kb, scb, ksz, rsqrtf((float)ksz));
    attn_softmax_q<<<(NEP * SEQL + 255) / 256, 256, 0, stream>>>(scb, atb);
    const long long n = (long long)TOTAL * (C + vsz);
    attn_out_concat<<<(int)((n + 255) / 256), 256, 0, stream>>>(xin, atb, vb, xout, C, vsz);
  };

  // ---- temporal-conv stack (5 dilated gated blocks, +128 ch each) ----
  auto tc_stage = [&](float*& cur, float*& alt, int& C, int pidx) {
    for (int i = 0; i < 5; ++i) {
      const int d = 2 << i;                    // 2^(i+1)
      const int Kp2 = al32(2 * C);
      const int pb = pidx + 4 * i;
      {
        const long long n = (long long)128 * Kp2;
        tc_repack_f16<<<(int)((n + 255) / 256), 256, 0, stream>>>(P[pb + 0], Bt1h, C, Kp2);
        tc_repack_f16<<<(int)((n + 255) / 256), 256, 0, stream>>>(P[pb + 2], Bt2h, C, Kp2);
      }
      {
        const long long n = (long long)TOTAL * Kp2;
        tc_gather_f16<<<(int)((n + 255) / 256), 256, 0, stream>>>(cur, Agh, C, Kp2, d);
      }
      gemm(Agh, Bt1h, P[pb + 1], yf, 128, Kp2);
      gemm(Agh, Bt2h, P[pb + 3], yg, 128, Kp2);
      {
        const long long n = (long long)TOTAL * (C + 128);
        tc_gate_concat<<<(int)((n + 255) / 256), 256, 0, stream>>>(cur, yf, yg, alt, C);
      }
      float* t = cur; cur = alt; alt = t;
      C += 128;
    }
  };

  float* cur = xA; float* alt = xB;
  int C = 69;
  attention(x0, C, 64, 32, 18, cur);  C = 101;             // att1
  tc_stage(cur, alt, C, 24);                                // tc1 -> 741
  attention(cur, C, 256, 128, 44, alt);
  { float* t = cur; cur = alt; alt = t; } C = 869;          // att2
  tc_stage(cur, alt, C, 50);                                // tc2 -> 1509
  attention(cur, C, 512, 256, 70, alt);
  { float* t = cur; cur = alt; alt = t; } C = 1765;         // att3

  // ---- final FC: [1664,1765] x [5,1765]^T + b -> d_out [64,26,5] ----
  {
    const int Cp = al32(C);
    cvtpad(cur, xh, TOTAL, C, Cp);
    cvtpad(P[76], wbh, 5, C, Cp);
    gemm(xh, wbh, P[77], (float*)d_out, 5, Cp);
  }

  (void)in_sizes; (void)n_in; (void)out_size; (void)ws_size;
}
